// GATv2Conv_Kmer_Classifier_57157424775866
// MI455X (gfx1250) — compile-verified, matrix operand order checked by reading, not run on hardware
//
#include <hip/hip_runtime.h>
#include <hip/hip_bf16.h>

// ---------------- problem constants ----------------
#define NNODE 10000
#define MPAD  10048           // 157 * 64 : padded row count for 64-row GEMM tiles
#define NEDGE 160000
#define ETOT  170000          // edges + self loops
#define F_IN  128
#define EMB   512
#define HID   768
#define HEADS 4
#define CH    192             // HID / HEADS

typedef __attribute__((ext_vector_type(16))) __bf16 v16bf;
typedef __attribute__((ext_vector_type(8)))  float  v8f;
typedef __attribute__((ext_vector_type(4)))  unsigned int v4u;
typedef __attribute__((ext_vector_type(8)))  int v8i;
typedef __attribute__((ext_vector_type(4)))  int v4i;

static __device__ __forceinline__ unsigned f2bf(float x) {
    unsigned u = __builtin_bit_cast(unsigned, x);
    return (u + 0x7FFFu + ((u >> 16) & 1u)) >> 16;   // round-to-nearest-even
}

// monotone float <-> ordered-uint for atomicMax on floats
static __device__ __forceinline__ unsigned ford(float x) {
    unsigned u = __builtin_bit_cast(unsigned, x);
    return u ^ ((u >> 31) ? 0xFFFFFFFFu : 0x80000000u);
}
static __device__ __forceinline__ float iford(unsigned o) {
    unsigned u = o ^ ((o >> 31) ? 0x80000000u : 0xFFFFFFFFu);
    return __builtin_bit_cast(float, u);
}

// Load a 16x32 bf16 fragment from an LDS tile with 64-element row stride.
// lane r=(lane&15), half=lane>>4 ; elems 0..7 <- K=half*8..+7, 8..15 <- K=16+half*8..+7
static __device__ __forceinline__ v16bf frag16x32s(const unsigned short* s, int lane) {
    int r = lane & 15, half = (lane >> 4) & 1;
    const unsigned short* p = s + r * 64 + half * 8;
    union { uint4 q[2]; v16bf v; } u;
    u.q[0] = *(const uint4*)(p);
    u.q[1] = *(const uint4*)(p + 16);
    return u.v;
}

// ---- Tensor Data Mover: DMA a 128(row) x 64(K-elem) bf16 tile into LDS ----
// gptr points at tile start inside Wt[N x K]; row stride = K elements (2B each).
static __device__ __forceinline__ void tdm_load_w(const unsigned short* gptr,
                                                  unsigned ldsOff, int K) {
    unsigned long long ga = (unsigned long long)(size_t)gptr;
    v4u g0;
    g0[0] = 1u;                                   // count=1 (valid user descriptor)
    g0[1] = ldsOff;                               // lds_addr (bytes)
    g0[2] = (unsigned)ga;                         // global_addr[31:0]
    g0[3] = (unsigned)((ga >> 32) & 0x01FFFFFFu)  // global_addr[56:32]
          | (2u << 30);                           // type = 2 ("image")
    unsigned u1[8];
    u1[0] = 1u << 16;                             // data_size = 1 -> 2 bytes
    u1[1] = 64u << 16;                            // tensor_dim0[15:0] = 64   (bits 63:48)
    u1[2] = 128u << 16;                           // dim0 hi=0 ; tensor_dim1[15:0]=128
    u1[3] = 64u << 16;                            // dim1 hi=0 ; tile_dim0 = 64
    u1[4] = 128u;                                 // tile_dim1 = 128 ; tile_dim2 = 0
    u1[5] = (unsigned)K;                          // tensor_dim0_stride[31:0] = K
    u1[6] = 0u;                                   // stride hi / dim1_stride = 0
    u1[7] = 0u;
    v8i g1;
    #pragma unroll
    for (int i = 0; i < 8; ++i) g1[i] = (int)u1[i];
    v4i z4 = {0, 0, 0, 0};
#if __has_include(<hip/amd_detail/amd_gfx1250_TDM.h>)
    v8i z8 = {0, 0, 0, 0, 0, 0, 0, 0};
    __builtin_amdgcn_tensor_load_to_lds(g0, g1, z4, z4, z8, 0);
#else
    __builtin_amdgcn_tensor_load_to_lds(g0, g1, z4, z4, 0);
#endif
}

// Stage one K-step: TDM pulls the 128x64 W tile; waves convert the 64x64 A tile.
static __device__ __forceinline__ void stage_tile(const float* __restrict__ A,
                                                  const unsigned short* __restrict__ Wt,
                                                  unsigned short* As, unsigned short* Ws,
                                                  int m0, int nb0, int K, int k0,
                                                  int tid, int wv) {
    if (wv == 0)
        tdm_load_w(Wt + (size_t)nb0 * K + k0, (unsigned)(size_t)(void*)Ws, K);
    #pragma unroll
    for (int t = 0; t < 4; ++t) {
        int j = tid + t * 256;                 // 0..1023 (uint2 granules)
        int row = j >> 4, c4 = (j & 15) * 4;
        const float4 v = *(const float4*)(A + (size_t)(m0 + row) * K + k0 + c4);
        uint2 p;
        p.x = f2bf(v.x) | (f2bf(v.y) << 16);
        p.y = f2bf(v.z) | (f2bf(v.w) << 16);
        ((uint2*)As)[j] = p;
    }
}

static __device__ __forceinline__ void compute_tile(const unsigned short* As,
                                                    const unsigned short* Ws,
                                                    int wm, int wn, int lane,
                                                    v8f& acc00, v8f& acc01,
                                                    v8f& acc10, v8f& acc11) {
    #pragma unroll
    for (int kc = 0; kc < 2; ++kc) {
        v16bf a0 = frag16x32s(As + (wm * 32)      * 64 + kc * 32, lane);
        v16bf a1 = frag16x32s(As + (wm * 32 + 16) * 64 + kc * 32, lane);
        v16bf b0 = frag16x32s(Ws + (wn * 32)      * 64 + kc * 32, lane);
        v16bf b1 = frag16x32s(Ws + (wn * 32 + 16) * 64 + kc * 32, lane);
        acc00 = __builtin_amdgcn_wmma_f32_16x16x32_bf16(false, a0, false, b0, (short)0, acc00, false, false);
        acc01 = __builtin_amdgcn_wmma_f32_16x16x32_bf16(false, a0, false, b1, (short)0, acc01, false, false);
        acc10 = __builtin_amdgcn_wmma_f32_16x16x32_bf16(false, a1, false, b0, (short)0, acc10, false, false);
        acc11 = __builtin_amdgcn_wmma_f32_16x16x32_bf16(false, a1, false, b1, (short)0, acc11, false, false);
    }
}

// ---------------- small utility kernels ----------------
__global__ void copy_f32(const float* __restrict__ in, float* __restrict__ out, int n) {
    int i = blockIdx.x * blockDim.x + threadIdx.x;
    if (i < n) out[i] = in[i];
}

__global__ void conv_wt(const float* __restrict__ w, unsigned short* __restrict__ wt,
                        int K, int Ncols) {
    int i = blockIdx.x * blockDim.x + threadIdx.x;
    if (i >= K * Ncols) return;
    int k = i / Ncols, n = i % Ncols;
    wt[(size_t)n * K + k] = (unsigned short)f2bf(w[i]);
}

__global__ void bnprep(const float* __restrict__ g, const float* __restrict__ b,
                       const float* __restrict__ m, const float* __restrict__ v,
                       float* __restrict__ scale, float* __restrict__ shift, int d) {
    int i = blockIdx.x * blockDim.x + threadIdx.x;
    if (i >= d) return;
    float s = g[i] * rsqrtf(v[i] + 1e-5f);
    scale[i] = s;
    shift[i] = b[i] - m[i] * s;
}

// ---------------- WMMA GEMM: C[M x N] = A[M x K](f32) @ Wt[N x K](bf16)^T ----------------
// 256 threads = 8 waves (2 x 4). Block tile 64(M) x 128(N); wave tile 32x32.
// K-step 64 (K % 64 == 0 for all layers). Double-buffered LDS: the Tensor Data
// Mover streams the next W tile and the waves stage the next A tile while the
// current K-step's 8 WMMAs per wave execute. One barrier per K-step.
__global__ __launch_bounds__(256)
void gemm_bf16(const float* __restrict__ A, const unsigned short* __restrict__ Wt,
               float* __restrict__ C, int K, int Ncols,
               const float* __restrict__ bias,
               const float* __restrict__ scale, const float* __restrict__ shift,
               int leaky) {
    __shared__ __align__(16) unsigned short As[2][64 * 64];     // 2 x 8 KB
    __shared__ __align__(16) unsigned short Ws[2][128 * 64];    // 2 x 16 KB

    const int tid  = threadIdx.x;
    const int lane = tid & 31;
    const int wv   = tid >> 5;
    const int wm   = wv >> 2;            // 0..1
    const int wn   = wv & 3;             // 0..3
    const int m0   = blockIdx.x * 64;
    const int nb0  = blockIdx.y * 128;

    v8f acc00 = {}, acc01 = {}, acc10 = {}, acc11 = {};

    const int ksteps = K >> 6;
    stage_tile(A, Wt, As[0], Ws[0], m0, nb0, K, 0, tid, wv);
    if (wv == 0) __builtin_amdgcn_s_wait_tensorcnt(0);
    __syncthreads();

    for (int kt = 0; kt < ksteps; ++kt) {
        const int cur = kt & 1;
        if (kt + 1 < ksteps)
            stage_tile(A, Wt, As[cur ^ 1], Ws[cur ^ 1], m0, nb0, K, (kt + 1) * 64, tid, wv);
        compute_tile(As[cur], Ws[cur], wm, wn, lane, acc00, acc01, acc10, acc11);
        if (wv == 0) __builtin_amdgcn_s_wait_tensorcnt(0);
        __syncthreads();
    }

    // ---- epilogue: D layout: VGPR r -> M = r + half*8 ; N = lane&15 ----
    const int half = lane >> 4;
    const int nl   = lane & 15;
    v8f accs[2][2] = {{acc00, acc01}, {acc10, acc11}};
    #pragma unroll
    for (int sm = 0; sm < 2; ++sm) {
        #pragma unroll
        for (int sn = 0; sn < 2; ++sn) {
            const int col = nb0 + wn * 32 + sn * 16 + nl;
            const float bsv = bias  ? bias[col]  : 0.f;
            const float sc  = scale ? scale[col] : 1.f;
            const float sh  = shift ? shift[col] : 0.f;
            const int rb = m0 + wm * 32 + sm * 16 + half * 8;
            #pragma unroll
            for (int r = 0; r < 8; ++r) {
                float v = accs[sm][sn][r] + bsv;
                v = v * sc + sh;
                if (leaky) v = (v >= 0.f) ? v : 0.01f * v;
                C[(size_t)(rb + r) * Ncols + col] = v;
            }
        }
    }
}

// ---------------- edge score: wave per edge; score[e][h] + segment max ----------------
__global__ __launch_bounds__(256)
void edge_score(const float* __restrict__ xl, const float* __restrict__ xr,
                const int* __restrict__ ei, const float* __restrict__ att,
                float* __restrict__ score, unsigned* __restrict__ mbuf) {
    int wid = blockIdx.x * (blockDim.x >> 5) + (threadIdx.x >> 5);
    if (wid >= ETOT) return;
    int lane = threadIdx.x & 31;
    int src, dst;
    if (wid < NEDGE) { src = ei[wid]; dst = ei[NEDGE + wid]; }
    else             { src = dst = wid - NEDGE; }

    const float* pl = xl + (size_t)src * HID;
    const float* pr = xr + (size_t)dst * HID;
    float acc[HEADS] = {0.f, 0.f, 0.f, 0.f};
    #pragma unroll
    for (int j = 0; j < CH / 32; ++j) {
        int c = j * 32 + lane;
        #pragma unroll
        for (int h = 0; h < HEADS; ++h) {
            int f = h * CH + c;
            float v = pl[f] + pr[f];
            v = (v >= 0.f) ? v : 0.2f * v;     // GATv2 internal leaky
            acc[h] += v * att[f];
        }
    }
    #pragma unroll
    for (int h = 0; h < HEADS; ++h) {
        float s = acc[h];
        #pragma unroll
        for (int off = 16; off > 0; off >>= 1) s += __shfl_xor(s, off, 32);
        if (lane == 0) {
            score[(size_t)wid * HEADS + h] = s;
            atomicMax(&mbuf[dst * HEADS + h], ford(s));
        }
    }
}

// ---------------- exp + denom accumulation ----------------
__global__ void edge_exp(float* __restrict__ score, const int* __restrict__ ei,
                         const unsigned* __restrict__ mbuf, float* __restrict__ denom) {
    int t = blockIdx.x * blockDim.x + threadIdx.x;
    if (t >= ETOT * HEADS) return;
    int e = t >> 2, h = t & 3;
    int dst = (e < NEDGE) ? ei[NEDGE + e] : (e - NEDGE);
    float mv = iford(mbuf[dst * HEADS + h]);
    float a = __expf(score[t] - mv);
    score[t] = a;
    atomicAdd(&denom[dst * HEADS + h], a);
}

// ---------------- weighted scatter-add: out[dst] += alpha * xl[src] ----------------
__global__ __launch_bounds__(256)
void edge_agg(const float* __restrict__ xl, const float* __restrict__ score,
              const float* __restrict__ denom, const int* __restrict__ ei,
              float* __restrict__ out) {
    int wid = blockIdx.x * (blockDim.x >> 5) + (threadIdx.x >> 5);
    if (wid >= ETOT) return;
    int lane = threadIdx.x & 31;
    int src, dst;
    if (wid < NEDGE) { src = ei[wid]; dst = ei[NEDGE + wid]; }
    else             { src = dst = wid - NEDGE; }

    float alpha[HEADS];
    #pragma unroll
    for (int h = 0; h < HEADS; ++h)
        alpha[h] = score[(size_t)wid * HEADS + h] / (denom[dst * HEADS + h] + 1e-16f);

    const float* p = xl + (size_t)src * HID;
    float* o = out + (size_t)dst * HID;
    for (int f = lane; f < HID; f += 32) {
        atomicAdd(&o[f], alpha[f / CH] * p[f]);
    }
}

// ---------------- post-GAT: out = leaky(bn(in + bias)) + resid ----------------
__global__ void post_gat(const float* __restrict__ in, const float* __restrict__ bias,
                         const float* __restrict__ scale, const float* __restrict__ shift,
                         const float* __restrict__ resid, float* __restrict__ out,
                         int total, int cols) {
    int i = blockIdx.x * blockDim.x + threadIdx.x;
    if (i >= total) return;
    int c = i % cols;
    float v = in[i] + bias[c];
    v = v * scale[c] + shift[c];
    v = (v >= 0.f) ? v : 0.01f * v;
    v += resid[i];
    out[i] = v;
}

// ---------------- final 384 -> 1 projection: wave per node ----------------
__global__ __launch_bounds__(256)
void lin_final(const float* __restrict__ h, const float* __restrict__ w,
               const float* __restrict__ b, float* __restrict__ out, int nodes, int K) {
    int wid = blockIdx.x * (blockDim.x >> 5) + (threadIdx.x >> 5);
    if (wid >= nodes) return;
    int lane = threadIdx.x & 31;
    const float* p = h + (size_t)wid * K;
    float s = 0.f;
    for (int c = lane; c < K; c += 32) s += p[c] * w[c];
    #pragma unroll
    for (int off = 16; off > 0; off >>= 1) s += __shfl_xor(s, off, 32);
    if (lane == 0) out[wid] = s + b[0];
}

static inline int cdiv(int a, int b) { return (a + b - 1) / b; }

extern "C" void kernel_launch(void* const* d_in, const int* in_sizes, int n_in,
                              void* d_out, int out_size, void* d_ws, size_t ws_size,
                              hipStream_t stream) {
    (void)in_sizes; (void)n_in; (void)out_size; (void)ws_size;

    // -------- inputs (setup_inputs dict order, params flattened) --------
    const float* x     = (const float*)d_in[0];
    const int*   ei    = (const int*)d_in[1];
    const float* pre1w = (const float*)d_in[2];  const float* pre1b = (const float*)d_in[3];
    const float* bn1g  = (const float*)d_in[4];  const float* bn1b  = (const float*)d_in[5];
    const float* bn1m  = (const float*)d_in[6];  const float* bn1v  = (const float*)d_in[7];
    const float* pre2w = (const float*)d_in[8];  const float* pre2b = (const float*)d_in[9];
    const float* bn2g  = (const float*)d_in[10]; const float* bn2b  = (const float*)d_in[11];
    const float* bn2m  = (const float*)d_in[12]; const float* bn2v  = (const float*)d_in[13];
    const float* resw  = (const float*)d_in[14]; const float* resb  = (const float*)d_in[15];
    const float* g0wl  = (const float*)d_in[16]; const float* g0wr  = (const float*)d_in[17];
    const float* g0att = (const float*)d_in[18]; const float* g0bias= (const float*)d_in[19];
    const float* n0g   = (const float*)d_in[20]; const float* n0b   = (const float*)d_in[21];
    const float* n0m   = (const float*)d_in[22]; const float* n0v   = (const float*)d_in[23];
    const float* g1wl  = (const float*)d_in[24]; const float* g1wr  = (const float*)d_in[25];
    const float* g1att = (const float*)d_in[26]; const float* g1bias= (const float*)d_in[27];
    const float* n1g   = (const float*)d_in[28]; const float* n1b   = (const float*)d_in[29];
    const float* n1m   = (const float*)d_in[30]; const float* n1v   = (const float*)d_in[31];
    const float* l1w   = (const float*)d_in[32]; const float* l1b   = (const float*)d_in[33];
    const float* l2w   = (const float*)d_in[34]; const float* l2b   = (const float*)d_in[35];
    const float* l3w   = (const float*)d_in[36]; const float* l3b   = (const float*)d_in[37];

    // -------- workspace carving --------
    char* wsp = (char*)d_ws;
    auto alloc = [&](size_t bytes) -> void* {
        void* p = (void*)wsp;
        wsp += (bytes + 255) & ~(size_t)255;
        return p;
    };
    unsigned short* wt_pre1 = (unsigned short*)alloc((size_t)EMB * F_IN * 2);
    unsigned short* wt_pre2 = (unsigned short*)alloc((size_t)EMB * EMB * 2);
    unsigned short* wt_res  = (unsigned short*)alloc((size_t)HID * EMB * 2);
    unsigned short* wt_g0l  = (unsigned short*)alloc((size_t)HID * EMB * 2);
    unsigned short* wt_g0r  = (unsigned short*)alloc((size_t)HID * EMB * 2);
    unsigned short* wt_g1l  = (unsigned short*)alloc((size_t)HID * HID * 2);
    unsigned short* wt_g1r  = (unsigned short*)alloc((size_t)HID * HID * 2);
    unsigned short* wt_l1   = (unsigned short*)alloc((size_t)HID * HID * 2);
    unsigned short* wt_l2   = (unsigned short*)alloc((size_t)(HID/2) * HID * 2);
    float* s1  = (float*)alloc(EMB * 4);  float* sh1  = (float*)alloc(EMB * 4);
    float* s2  = (float*)alloc(EMB * 4);  float* sh2  = (float*)alloc(EMB * 4);
    float* s0n = (float*)alloc(HID * 4);  float* sh0n = (float*)alloc(HID * 4);
    float* s1n = (float*)alloc(HID * 4);  float* sh1n = (float*)alloc(HID * 4);
    float* xpad = (float*)alloc((size_t)MPAD * F_IN * 4);
    float* hA   = (float*)alloc((size_t)MPAD * EMB * 4);
    float* hB   = (float*)alloc((size_t)MPAD * EMB * 4);
    float* xl   = (float*)alloc((size_t)MPAD * HID * 4);
    float* xr   = (float*)alloc((size_t)MPAD * HID * 4);
    float* resv = (float*)alloc((size_t)MPAD * HID * 4);
    float* agg  = (float*)alloc((size_t)MPAD * HID * 4);
    float* hC   = (float*)alloc((size_t)MPAD * HID * 4);
    float*    score = (float*)alloc((size_t)ETOT * HEADS * 4);
    unsigned* mbuf  = (unsigned*)alloc((size_t)NNODE * HEADS * 4);
    float*    denom = (float*)alloc((size_t)NNODE * HEADS * 4);

    const dim3 blk(256);
    const int MT = MPAD / 64;   // 157

    // -------- weight conversion (f32 -> transposed bf16) + BN prep --------
    conv_wt<<<cdiv(F_IN * EMB, 256), blk, 0, stream>>>(pre1w, wt_pre1, F_IN, EMB);
    conv_wt<<<cdiv(EMB * EMB, 256),  blk, 0, stream>>>(pre2w, wt_pre2, EMB, EMB);
    conv_wt<<<cdiv(EMB * HID, 256),  blk, 0, stream>>>(resw,  wt_res,  EMB, HID);
    conv_wt<<<cdiv(EMB * HID, 256),  blk, 0, stream>>>(g0wl,  wt_g0l,  EMB, HID);
    conv_wt<<<cdiv(EMB * HID, 256),  blk, 0, stream>>>(g0wr,  wt_g0r,  EMB, HID);
    conv_wt<<<cdiv(HID * HID, 256),  blk, 0, stream>>>(g1wl,  wt_g1l,  HID, HID);
    conv_wt<<<cdiv(HID * HID, 256),  blk, 0, stream>>>(g1wr,  wt_g1r,  HID, HID);
    conv_wt<<<cdiv(HID * HID, 256),  blk, 0, stream>>>(l1w,   wt_l1,   HID, HID);
    conv_wt<<<cdiv(HID * (HID/2), 256), blk, 0, stream>>>(l2w, wt_l2,  HID, HID/2);

    bnprep<<<cdiv(EMB, 256), blk, 0, stream>>>(bn1g, bn1b, bn1m, bn1v, s1,  sh1,  EMB);
    bnprep<<<cdiv(EMB, 256), blk, 0, stream>>>(bn2g, bn2b, bn2m, bn2v, s2,  sh2,  EMB);
    bnprep<<<cdiv(HID, 256), blk, 0, stream>>>(n0g,  n0b,  n0m,  n0v,  s0n, sh0n, HID);
    bnprep<<<cdiv(HID, 256), blk, 0, stream>>>(n1g,  n1b,  n1m,  n1v,  s1n, sh1n, HID);

    // -------- pre-MLP --------
    copy_f32<<<cdiv(NNODE * F_IN, 256), blk, 0, stream>>>(x, xpad, NNODE * F_IN);
    gemm_bf16<<<dim3(MT, EMB / 128), blk, 0, stream>>>(xpad, wt_pre1, hA, F_IN, EMB, pre1b, s1, sh1, 1);
    gemm_bf16<<<dim3(MT, EMB / 128), blk, 0, stream>>>(hA,   wt_pre2, hB, EMB,  EMB, pre2b, s2, sh2, 1);

    // residual projection + GAT0 transforms
    gemm_bf16<<<dim3(MT, HID / 128), blk, 0, stream>>>(hB, wt_res, resv, EMB, HID, resb, nullptr, nullptr, 0);
    gemm_bf16<<<dim3(MT, HID / 128), blk, 0, stream>>>(hB, wt_g0l, xl,   EMB, HID, nullptr, nullptr, nullptr, 0);
    gemm_bf16<<<dim3(MT, HID / 128), blk, 0, stream>>>(hB, wt_g0r, xr,   EMB, HID, nullptr, nullptr, nullptr, 0);

    // -------- GAT layer 0 --------
    hipMemsetAsync(agg,   0, (size_t)NNODE * HID * 4,   stream);
    hipMemsetAsync(mbuf,  0, (size_t)NNODE * HEADS * 4, stream);
    hipMemsetAsync(denom, 0, (size_t)NNODE * HEADS * 4, stream);
    edge_score<<<cdiv(ETOT, 8), blk, 0, stream>>>(xl, xr, ei, g0att, score, mbuf);
    edge_exp<<<cdiv(ETOT * HEADS, 256), blk, 0, stream>>>(score, ei, mbuf, denom);
    edge_agg<<<cdiv(ETOT, 8), blk, 0, stream>>>(xl, score, denom, ei, agg);
    post_gat<<<cdiv(NNODE * HID, 256), blk, 0, stream>>>(agg, g0bias, s0n, sh0n, resv, hC, NNODE * HID, HID);

    // -------- GAT layer 1 --------
    gemm_bf16<<<dim3(MT, HID / 128), blk, 0, stream>>>(hC, wt_g1l, xl, HID, HID, nullptr, nullptr, nullptr, 0);
    gemm_bf16<<<dim3(MT, HID / 128), blk, 0, stream>>>(hC, wt_g1r, xr, HID, HID, nullptr, nullptr, nullptr, 0);
    hipMemsetAsync(agg,   0, (size_t)NNODE * HID * 4,   stream);
    hipMemsetAsync(mbuf,  0, (size_t)NNODE * HEADS * 4, stream);
    hipMemsetAsync(denom, 0, (size_t)NNODE * HEADS * 4, stream);
    edge_score<<<cdiv(ETOT, 8), blk, 0, stream>>>(xl, xr, ei, g1att, score, mbuf);
    edge_exp<<<cdiv(ETOT * HEADS, 256), blk, 0, stream>>>(score, ei, mbuf, denom);
    edge_agg<<<cdiv(ETOT, 8), blk, 0, stream>>>(xl, score, denom, ei, agg);
    post_gat<<<cdiv(NNODE * HID, 256), blk, 0, stream>>>(agg, g1bias, s1n, sh1n, hC, hC, NNODE * HID, HID);

    // -------- output MLP --------
    gemm_bf16<<<dim3(MT, HID / 128),     blk, 0, stream>>>(hC, wt_l1, xl, HID, HID,     l1b, nullptr, nullptr, 1);
    gemm_bf16<<<dim3(MT, (HID/2) / 128), blk, 0, stream>>>(xl, wt_l2, xr, HID, HID / 2, l2b, nullptr, nullptr, 1);
    lin_final<<<cdiv(NNODE, 8), blk, 0, stream>>>(xr, l3w, l3b, (float*)d_out, NNODE, HID / 2);
}
